// PredictiveCodingAgent_13486197309663
// MI455X (gfx1250) — matrix operand order repaired and checked by reading, not compiled, not run on hardware
//
#include <hip/hip_runtime.h>
#include <hip/hip_bf16.h>
#include <math.h>

// ---------------------------------------------------------------------------
// PredictiveCodingAgent for MI455X (gfx1250, wave32, WMMA bf16)
// All f32->bf16 conversion hoisted into one-pass prep kernels; WMMA GEMM
// inner loops are pure b128 loads + v_wmma_f32_16x16x32_bf16.
// ---------------------------------------------------------------------------

typedef __attribute__((ext_vector_type(16))) __bf16 v16bf;
typedef __attribute__((ext_vector_type(8)))  float  v8f;

#define BQ    1024
#define MV    65536
#define DD    384
#define HH    384
#define TOPK  5
#define SEQ   6
#define CEPS  1e-8f
#define MCHUNK 512
#define NCH   (MV / MCHUNK)   // 128

union FragAB { v16bf v; unsigned short us[16]; uint4 q[2]; };
union Acc8   { v8f v; float f[8]; };

__device__ inline unsigned short f2bf(float f) {
    unsigned int u = __float_as_uint(f);
    u += 0x7fffu + ((u >> 16) & 1u);          // round-to-nearest-even
    return (unsigned short)(u >> 16);
}

__device__ inline void ins5(float v, int id, float* tv, int* ti) {
    if (v > tv[4]) {
        tv[4] = v; ti[4] = id;
        #pragma unroll
        for (int j = 4; j > 0; --j) {
            if (tv[j] > tv[j - 1]) {
                float tf = tv[j]; tv[j] = tv[j - 1]; tv[j - 1] = tf;
                int   td = ti[j]; ti[j] = ti[j - 1]; ti[j - 1] = td;
            }
        }
    }
}

__device__ inline uint4 ldq(const unsigned short* p) {
    return *(const uint4*)(const void*)p;
}

// --------------------------- prep kernels ----------------------------------

// bulk f32 -> bf16 (n must be a multiple of 4)
__global__ void cvt_bf16_kernel(const float* __restrict__ src,
                                unsigned short* __restrict__ dst, int n4) {
    const int i = blockIdx.x * blockDim.x + threadIdx.x;
    if (i < n4) {
        float4 v = ((const float4*)src)[i];
        ushort4 o;
        o.x = f2bf(v.x); o.y = f2bf(v.y); o.z = f2bf(v.z); o.w = f2bf(v.w);
        ((ushort4*)dst)[i] = o;
    }
}

__global__ void center_kernel(const float* __restrict__ sw, float* __restrict__ center) {
    __shared__ float sx[64], sy[64];
    const int t = threadIdx.x;
    float ax = 0.f, ay = 0.f;
    for (int i = t; i < HH; i += 64) { ax += sw[i * 2]; ay += sw[i * 2 + 1]; }
    sx[t] = ax; sy[t] = ay;
    __syncthreads();
    for (int s = 32; s > 0; s >>= 1) {
        if (t < s) { sx[t] += sx[t + s]; sy[t] += sy[t + s]; }
        __syncthreads();
    }
    if (t == 0) { center[0] = sx[0] / (float)HH; center[1] = sy[0] / (float)HH; }
}

__global__ void mem_prep_kernel(const float* __restrict__ mem, const float* __restrict__ coords,
                                const float* __restrict__ center,
                                float* __restrict__ mn, float* __restrict__ act) {
    const int m    = blockIdx.x * 8 + (threadIdx.x >> 5);
    const int lane = threadIdx.x & 31;
    const float* row = mem + (size_t)m * DD;
    float s = 0.f;
    #pragma unroll
    for (int j = 0; j < DD / 32; ++j) { float v = row[lane + 32 * j]; s += v * v; }
    #pragma unroll
    for (int off = 16; off > 0; off >>= 1) s += __shfl_xor(s, off, 32);
    if (lane == 0) {
        mn[m] = sqrtf(s);
        float dx = coords[m * 2]     - center[0];
        float dy = coords[m * 2 + 1] - center[1];
        act[m] = 1.0f / (1.0f + sqrtf(dx * dx + dy * dy));
    }
}

__global__ void query_prep_kernel(const float* __restrict__ query,
                                  float* __restrict__ qn, unsigned short* __restrict__ qus) {
    const int b    = blockIdx.x * 8 + (threadIdx.x >> 5);
    const int lane = threadIdx.x & 31;
    const float* row = query + (size_t)b * DD;
    float s = 0.f;
    #pragma unroll
    for (int j = 0; j < DD / 32; ++j) {
        float v = row[lane + 32 * j];
        qus[(size_t)b * DD + lane + 32 * j] = f2bf(v);
        s += v * v;
    }
    #pragma unroll
    for (int off = 16; off > 0; off >>= 1) s += __shfl_xor(s, off, 32);
    if (lane == 0) qn[b] = sqrtf(s);
}

__global__ void zero_us_kernel(unsigned short* __restrict__ p, int n) {
    const int i = blockIdx.x * blockDim.x + threadIdx.x;
    if (i < n) p[i] = 0;
}

// ------------------ big GEMM (cosine sim) + fused chunk top-k --------------

__global__ __launch_bounds__(256) void sim_topk_kernel(
    const unsigned short* __restrict__ qbf, const unsigned short* __restrict__ mbf,
    const float* __restrict__ qn, const float* __restrict__ mn,
    const float* __restrict__ act,
    float* __restrict__ pval, int* __restrict__ pidx)
{
    __shared__ float sim_s[16 * MCHUNK];       // 32 KB
    __shared__ float cv_s[16 * 16 * TOPK];     // 5 KB
    __shared__ int   ci_s[16 * 16 * TOPK];     // 5 KB

    const int mc = blockIdx.x;                 // mem chunk  (0..127)
    const int qt = blockIdx.y;                 // query tile (0..63)
    const int q0 = qt * 16;
    const int m0 = mc * MCHUNK;

    const int lane  = threadIdx.x & 31;
    const int wv    = threadIdx.x >> 5;
    const int r     = lane & 15;
    const int hi    = lane >> 4;
    const int khalf = hi * 8;

    // phase 1: WMMA bf16, accumulate f32; 16 queries x 512 mem rows
    const unsigned short* qrow = qbf + (size_t)(q0 + r) * DD;
    FragAB afrag[12];
    #pragma unroll
    for (int ks = 0; ks < 12; ++ks) {
        const int kb = ks * 32;
        afrag[ks].q[0] = ldq(qrow + kb + khalf);
        afrag[ks].q[1] = ldq(qrow + kb + 16 + khalf);
    }

    float qn8[8];
    #pragma unroll
    for (int vr = 0; vr < 8; ++vr) qn8[vr] = qn[q0 + vr + hi * 8];

    for (int pass = 0; pass < MCHUNK / 128; ++pass) {
        const int tloc = (pass * 8 + wv) * 16;            // local m tile base
        const unsigned short* mrow = mbf + (size_t)(m0 + tloc + r) * DD;
        // prefetch next pass's B rows (global_prefetch_b8)
        if (pass + 1 < MCHUNK / 128)
            __builtin_prefetch(mrow + (size_t)128 * DD, 0, 1);

        Acc8 c; c.v = (v8f){0.f, 0.f, 0.f, 0.f, 0.f, 0.f, 0.f, 0.f};
        #pragma unroll
        for (int ks = 0; ks < 12; ++ks) {
            const int kb = ks * 32;
            FragAB b;
            b.q[0] = ldq(mrow + kb + hi * 8);
            b.q[1] = ldq(mrow + kb + 16 + hi * 8);
            c.v = __builtin_amdgcn_wmma_f32_16x16x32_bf16(
                false, afrag[ks].v, false, b.v, (short)0, c.v, false, false);
        }
        const int   lc  = tloc + r;                       // local column
        const float mnv = mn[m0 + lc];
        const float av  = act[m0 + lc];
        #pragma unroll
        for (int vr = 0; vr < 8; ++vr) {
            const float denom = fmaxf(qn8[vr] * mnv, CEPS);
            sim_s[(vr + hi * 8) * MCHUNK + lc] = c.f[vr] / denom + av;
        }
    }
    __syncthreads();

    // phase 2: per-row, per-segment top-5 over 32 columns each
    {
        const int row = threadIdx.x >> 4;
        const int seg = threadIdx.x & 15;
        float tv[5]; int ti[5];
        #pragma unroll
        for (int i = 0; i < 5; ++i) { tv[i] = -INFINITY; ti[i] = 0; }
        const int c0 = seg * (MCHUNK / 16);
        for (int cc = 0; cc < MCHUNK / 16; ++cc)
            ins5(sim_s[row * MCHUNK + c0 + cc], c0 + cc, tv, ti);
        #pragma unroll
        for (int i = 0; i < 5; ++i) {
            cv_s[(row * 16 + seg) * 5 + i] = tv[i];
            ci_s[(row * 16 + seg) * 5 + i] = ti[i];
        }
    }
    __syncthreads();

    // phase 3: merge 16 segments -> chunk top-5, write partial candidates
    if (threadIdx.x < 16) {
        const int row = threadIdx.x;
        float tv[5]; int ti[5];
        #pragma unroll
        for (int i = 0; i < 5; ++i) { tv[i] = -INFINITY; ti[i] = 0; }
        for (int c = 0; c < 16 * 5; ++c)
            ins5(cv_s[row * 80 + c], ci_s[row * 80 + c], tv, ti);
        const size_t base = ((size_t)(q0 + row) * NCH + mc) * TOPK;
        #pragma unroll
        for (int i = 0; i < 5; ++i) {
            pval[base + i] = tv[i];
            pidx[base + i] = m0 + ti[i];
        }
    }
}

__global__ void topk_merge_kernel(const float* __restrict__ pval,
                                  const int* __restrict__ pidx, int* __restrict__ topk) {
    const int q = blockIdx.x * blockDim.x + threadIdx.x;
    if (q >= BQ) return;
    float tv[5]; int ti[5];
    #pragma unroll
    for (int i = 0; i < 5; ++i) { tv[i] = -INFINITY; ti[i] = 0; }
    const size_t base = (size_t)q * NCH * TOPK;
    for (int c = 0; c < NCH * TOPK; ++c) ins5(pval[base + c], pidx[base + c], tv, ti);
    #pragma unroll
    for (int i = 0; i < 5; ++i) topk[q * TOPK + i] = ti[i];
}

// ----------------- x_proj = [query; retrieved] @ W_ih.T + b_ih -------------

__global__ __launch_bounds__(256) void xproj_kernel(
    const unsigned short* __restrict__ qbf, const unsigned short* __restrict__ mbf,
    const int* __restrict__ topk, const unsigned short* __restrict__ wbf,
    const float* __restrict__ b_ih, float* __restrict__ xp)
{
    const int wg = blockIdx.x * 8 + (threadIdx.x >> 5);   // 0..9215
    const int rt = wg / (HH / 16);                        // row tile (0..383)
    const int ht = wg % (HH / 16);                        // h tile   (0..23)
    const int lane = threadIdx.x & 31;
    const int r = lane & 15, hi = lane >> 4, khalf = hi * 8;

    const int gr = rt * 16 + r;                           // global seq row
    const int b  = gr / SEQ;
    const int t  = gr % SEQ;
    const unsigned short* arow = (t == 0)
        ? (qbf + (size_t)b * DD)
        : (mbf + (size_t)topk[b * TOPK + (t - 1)] * DD);
    const unsigned short* wrow = wbf + (size_t)(ht * 16 + r) * DD;

    Acc8 c; c.v = (v8f){0.f, 0.f, 0.f, 0.f, 0.f, 0.f, 0.f, 0.f};
    #pragma unroll
    for (int ks = 0; ks < 12; ++ks) {
        const int kb = ks * 32;
        FragAB a, bb;
        a.q[0]  = ldq(arow + kb + khalf);
        a.q[1]  = ldq(arow + kb + 16 + khalf);
        bb.q[0] = ldq(wrow + kb + hi * 8);
        bb.q[1] = ldq(wrow + kb + 16 + hi * 8);
        c.v = __builtin_amdgcn_wmma_f32_16x16x32_bf16(
            false, a.v, false, bb.v, (short)0, c.v, false, false);
    }
    const int   hcol = ht * 16 + r;
    const float bias = b_ih[hcol];
    #pragma unroll
    for (int vr = 0; vr < 8; ++vr) {
        const int row = rt * 16 + vr + hi * 8;
        xp[(size_t)row * HH + hcol] = c.f[vr] + bias;
    }
}

// ------------- h = tanh(x_t + h @ W_hh.T + b_hh), double-buffered ----------

__global__ __launch_bounds__(256) void rnn_step_kernel(
    const float* __restrict__ xp, const unsigned short* __restrict__ wbf,
    const float* __restrict__ b_hh, const unsigned short* __restrict__ hin,
    unsigned short* __restrict__ hout, float* __restrict__ hf, int t)
{
    const int wg = blockIdx.x * 8 + (threadIdx.x >> 5);   // 0..1535
    const int bt = wg / (HH / 16);                        // batch tile (0..63)
    const int ht = wg % (HH / 16);
    const int lane = threadIdx.x & 31;
    const int r = lane & 15, hi = lane >> 4, khalf = hi * 8;

    const unsigned short* hrow = hin + (size_t)(bt * 16 + r) * HH;
    const unsigned short* wrow = wbf + (size_t)(ht * 16 + r) * DD;

    Acc8 c; c.v = (v8f){0.f, 0.f, 0.f, 0.f, 0.f, 0.f, 0.f, 0.f};
    #pragma unroll
    for (int ks = 0; ks < 12; ++ks) {
        const int kb = ks * 32;
        FragAB a, bb;
        a.q[0]  = ldq(hrow + kb + khalf);
        a.q[1]  = ldq(hrow + kb + 16 + khalf);
        bb.q[0] = ldq(wrow + kb + hi * 8);
        bb.q[1] = ldq(wrow + kb + 16 + hi * 8);
        c.v = __builtin_amdgcn_wmma_f32_16x16x32_bf16(
            false, a.v, false, bb.v, (short)0, c.v, false, false);
    }
    const int   hcol = ht * 16 + r;
    const float bh   = b_hh[hcol];
    #pragma unroll
    for (int vr = 0; vr < 8; ++vr) {
        const int brow = bt * 16 + vr + hi * 8;
        float v = tanhf(c.f[vr] + xp[((size_t)brow * SEQ + t) * HH + hcol] + bh);
        hf[(size_t)brow * HH + hcol]   = v;
        hout[(size_t)brow * HH + hcol] = f2bf(v);
    }
}

// ---------- out = sigmoid(q @ W_g.T + b_g) * h + (1-gate) * direct ---------

__global__ __launch_bounds__(256) void gate_out_kernel(
    const unsigned short* __restrict__ qbf, const unsigned short* __restrict__ wbf,
    const float* __restrict__ b_g, const float* __restrict__ hf,
    const float* __restrict__ xp, float* __restrict__ out)
{
    const int wg = blockIdx.x * 8 + (threadIdx.x >> 5);
    const int bt = wg / (HH / 16);
    const int ht = wg % (HH / 16);
    const int lane = threadIdx.x & 31;
    const int r = lane & 15, hi = lane >> 4, khalf = hi * 8;

    const unsigned short* qrow = qbf + (size_t)(bt * 16 + r) * DD;
    const unsigned short* wrow = wbf + (size_t)(ht * 16 + r) * DD;

    Acc8 c; c.v = (v8f){0.f, 0.f, 0.f, 0.f, 0.f, 0.f, 0.f, 0.f};
    #pragma unroll
    for (int ks = 0; ks < 12; ++ks) {
        const int kb = ks * 32;
        FragAB a, bb;
        a.q[0]  = ldq(qrow + kb + khalf);
        a.q[1]  = ldq(qrow + kb + 16 + khalf);
        bb.q[0] = ldq(wrow + kb + hi * 8);
        bb.q[1] = ldq(wrow + kb + 16 + hi * 8);
        c.v = __builtin_amdgcn_wmma_f32_16x16x32_bf16(
            false, a.v, false, bb.v, (short)0, c.v, false, false);
    }
    const int   hcol = ht * 16 + r;
    const float bg   = b_g[hcol];
    #pragma unroll
    for (int vr = 0; vr < 8; ++vr) {
        const int brow = bt * 16 + vr + hi * 8;
        float g   = 1.0f / (1.0f + expf(-(c.f[vr] + bg)));
        float hv  = hf[(size_t)brow * HH + hcol];
        float dir = xp[((size_t)brow * SEQ + 0) * HH + hcol];
        out[(size_t)brow * HH + hcol] = g * hv + (1.0f - g) * dir;
    }
}

// --------------------------------- launch ----------------------------------

extern "C" void kernel_launch(void* const* d_in, const int* in_sizes, int n_in,
                              void* d_out, int out_size, void* d_ws, size_t ws_size,
                              hipStream_t stream)
{
    const float* query  = (const float*)d_in[0];
    const float* mem    = (const float*)d_in[1];
    const float* coords = (const float*)d_in[2];
    const float* sw     = (const float*)d_in[3];
    const float* W_ih   = (const float*)d_in[4];
    const float* b_ih   = (const float*)d_in[5];
    const float* W_hh   = (const float*)d_in[6];
    const float* b_hh   = (const float*)d_in[7];
    const float* W_g    = (const float*)d_in[8];
    const float* b_g    = (const float*)d_in[9];
    float* out = (float*)d_out;

    char* ws = (char*)d_ws;
    size_t off = 0;
    auto take = [&](size_t bytes) -> char* {
        char* p = ws + off;
        off += (bytes + 255) & ~(size_t)255;
        return p;
    };

    float*          center = (float*)take(2 * sizeof(float));
    float*          qn     = (float*)take((size_t)BQ * sizeof(float));
    float*          mn     = (float*)take((size_t)MV * sizeof(float));
    float*          act    = (float*)take((size_t)MV * sizeof(float));
    float*          pval   = (float*)take((size_t)BQ * NCH * TOPK * sizeof(float));
    int*            pidx   = (int*)  take((size_t)BQ * NCH * TOPK * sizeof(int));
    int*            topk   = (int*)  take((size_t)BQ * TOPK * sizeof(int));
    float*          xp     = (float*)take((size_t)BQ * SEQ * HH * sizeof(float));
    float*          hf     = (float*)take((size_t)BQ * HH * sizeof(float));
    unsigned short* qus    = (unsigned short*)take((size_t)BQ * DD * sizeof(unsigned short));
    unsigned short* ha     = (unsigned short*)take((size_t)BQ * HH * sizeof(unsigned short));
    unsigned short* hb     = (unsigned short*)take((size_t)BQ * HH * sizeof(unsigned short));
    unsigned short* wihbf  = (unsigned short*)take((size_t)HH * DD * sizeof(unsigned short));
    unsigned short* whhbf  = (unsigned short*)take((size_t)HH * DD * sizeof(unsigned short));
    unsigned short* wgbf   = (unsigned short*)take((size_t)HH * DD * sizeof(unsigned short));
    unsigned short* membf  = (unsigned short*)take((size_t)MV * DD * sizeof(unsigned short));

    // ---- one-pass conversions & norms ----
    center_kernel<<<1, 64, 0, stream>>>(sw, center);
    mem_prep_kernel<<<MV / 8, 256, 0, stream>>>(mem, coords, center, mn, act);
    query_prep_kernel<<<BQ / 8, 256, 0, stream>>>(query, qn, qus);
    cvt_bf16_kernel<<<((MV * DD / 4) + 255) / 256, 256, 0, stream>>>(mem, membf, MV * DD / 4);
    cvt_bf16_kernel<<<((HH * DD / 4) + 255) / 256, 256, 0, stream>>>(W_ih, wihbf, HH * DD / 4);
    cvt_bf16_kernel<<<((HH * DD / 4) + 255) / 256, 256, 0, stream>>>(W_hh, whhbf, HH * DD / 4);
    cvt_bf16_kernel<<<((HH * DD / 4) + 255) / 256, 256, 0, stream>>>(W_g, wgbf, HH * DD / 4);

    // ---- similarity GEMM + top-k ----
    sim_topk_kernel<<<dim3(NCH, BQ / 16), 256, 0, stream>>>(qus, membf, qn, mn, act, pval, pidx);
    topk_merge_kernel<<<(BQ + 255) / 256, 256, 0, stream>>>(pval, pidx, topk);

    // ---- contextualize ----
    xproj_kernel<<<(BQ * SEQ / 16) * (HH / 16) / 8, 256, 0, stream>>>(qus, membf, topk, wihbf, b_ih, xp);

    zero_us_kernel<<<(BQ * HH + 255) / 256, 256, 0, stream>>>(ha, BQ * HH);
    unsigned short* hin = ha;
    unsigned short* hbuf = hb;
    for (int t = 0; t < SEQ; ++t) {
        rnn_step_kernel<<<(BQ / 16) * (HH / 16) / 8, 256, 0, stream>>>(xp, whhbf, b_hh, hin, hbuf, hf, t);
        unsigned short* tmp = hin; hin = hbuf; hbuf = tmp;
    }

    gate_out_kernel<<<(BQ / 16) * (HH / 16) / 8, 256, 0, stream>>>(qus, wgbf, b_g, hf, xp, out);

    (void)in_sizes; (void)n_in; (void)out_size; (void)ws_size;
}
